// UnifiedAdaptiveTNN_13280038879585
// MI455X (gfx1250) — matrix operand order repaired and checked by reading, not compiled, not run on hardware
//
#include <hip/hip_runtime.h>
#include <hip/hip_bf16.h>
#include <math.h>

// ---------------------------------------------------------------------------
// Types for CDNA5 WMMA (wave32): v16h = A/B fragment, v8f = C/D fragment
// ---------------------------------------------------------------------------
typedef __attribute__((ext_vector_type(4)))  unsigned int v4u;
typedef __attribute__((ext_vector_type(4)))  float        v4f;
typedef __attribute__((ext_vector_type(8)))  float        v8f;
typedef __attribute__((ext_vector_type(4)))  _Float16     v4h;
typedef __attribute__((ext_vector_type(16))) _Float16     v16h;

union HFrag { v4u u[2]; v16h h; };

#define DMODEL 1024
#define MROWS  2048   // B*S
#define NLAYER 2
#define NBLK   4

__device__ __forceinline__ float sigmoidf_(float x) { return 1.0f / (1.0f + expf(-x)); }
__device__ __forceinline__ float gelu_exact(float x) {
    return 0.5f * x * (1.0f + erff(x * 0.70710678118654752f));
}

// ---------------------------------------------------------------------------
// Tiny kernel: excitement weights excn[l][k] = exc/sum(exc), constant per call
// ---------------------------------------------------------------------------
__global__ void exc_kernel(const float* __restrict__ propW,
                           const float* __restrict__ propb,
                           float* __restrict__ excn) {
    if (threadIdx.x == 0) {
        const float nsig = 1.0f / (1.0f + expf(-0.70710678118654752f)); // NEIGHBOR_SIG
        for (int l = 0; l < NLAYER; ++l) {
            float e[NBLK]; float sum = 0.0f;
            for (int k = 0; k < NBLK; ++k) {
                float a = propb[l * NBLK + k];
                for (int j = 0; j < NBLK; ++j)
                    a += 0.5f * propW[(l * NBLK + k) * NBLK + j];
                e[k] = 0.6f * sigmoidf_(a) + 0.4f * nsig;
                sum += e[k];
            }
            for (int k = 0; k < NBLK; ++k)
                excn[l * NBLK + k] = e[k] / sum;
        }
    }
}

// ---------------------------------------------------------------------------
// Embedding gather + positional encoding: h[row,d] = emb[ids[row],d]+pos[s,d]
// ---------------------------------------------------------------------------
__global__ __launch_bounds__(256)
void embed_kernel(const int* __restrict__ ids, const float* __restrict__ emb,
                  const float* __restrict__ pos, float* __restrict__ h) {
    const int row = blockIdx.x;          // 0..2047
    const int s   = row & (1024 - 1);    // S = 1024
    const int t   = threadIdx.x;
    const int id  = ids[row];
    v4f e = *(const v4f*)(emb + (size_t)id * DMODEL + t * 4);
    v4f p = *(const v4f*)(pos + (size_t)s  * DMODEL + t * 4);
    v4f o; o.x = e.x + p.x; o.y = e.y + p.y; o.z = e.z + p.z; o.w = e.w + p.w;
    *(v4f*)(h + (size_t)row * DMODEL + t * 4) = o;
}

// ---------------------------------------------------------------------------
// Block reduction helper (256 threads)
// ---------------------------------------------------------------------------
__device__ __forceinline__ float block_sum(float v, float* red) {
    const int t = threadIdx.x;
    red[t] = v; __syncthreads();
    #pragma unroll
    for (int off = 128; off > 0; off >>= 1) {
        if (t < off) red[t] += red[t + off];
        __syncthreads();
    }
    float r = red[0];
    __syncthreads();
    return r;
}

// ---------------------------------------------------------------------------
// LayerNorm(src) * g + b  -> f16 output (one block per row of 1024)
// ---------------------------------------------------------------------------
__global__ __launch_bounds__(256)
void ln_to_f16_kernel(const float* __restrict__ src, const float* __restrict__ g,
                      const float* __restrict__ b, _Float16* __restrict__ dst) {
    __shared__ float red[256];
    const int row = blockIdx.x, t = threadIdx.x;
    v4f v = *(const v4f*)(src + (size_t)row * DMODEL + t * 4);
    float mu = block_sum(v.x + v.y + v.z + v.w, red) * (1.0f / DMODEL);
    v4f d; d.x = v.x - mu; d.y = v.y - mu; d.z = v.z - mu; d.w = v.w - mu;
    float var = block_sum(d.x*d.x + d.y*d.y + d.z*d.z + d.w*d.w, red) * (1.0f / DMODEL);
    float rstd = rsqrtf(var + 1e-5f);
    v4f gg = *(const v4f*)(g + t * 4);
    v4f bb = *(const v4f*)(b + t * 4);
    v4h o;
    o.x = (_Float16)(d.x * rstd * gg.x + bb.x);
    o.y = (_Float16)(d.y * rstd * gg.y + bb.y);
    o.z = (_Float16)(d.z * rstd * gg.z + bb.z);
    o.w = (_Float16)(d.w * rstd * gg.w + bb.w);
    *(v4h*)(dst + (size_t)row * DMODEL + t * 4) = o;
}

// ---------------------------------------------------------------------------
// h = LayerNorm(h + 0.3*(base+comb)) * g + b   (in place, f32)
// ---------------------------------------------------------------------------
__global__ __launch_bounds__(256)
void resid_ln_kernel(float* __restrict__ h, const float* __restrict__ base,
                     const float* __restrict__ comb, const float* __restrict__ g,
                     const float* __restrict__ b) {
    __shared__ float red[256];
    const int row = blockIdx.x, t = threadIdx.x;
    const size_t o4 = (size_t)row * DMODEL + t * 4;
    v4f hv = *(const v4f*)(h + o4);
    v4f bv = *(const v4f*)(base + o4);
    v4f cv = *(const v4f*)(comb + o4);
    v4f x;
    x.x = hv.x + 0.3f * (bv.x + cv.x);
    x.y = hv.y + 0.3f * (bv.y + cv.y);
    x.z = hv.z + 0.3f * (bv.z + cv.z);
    x.w = hv.w + 0.3f * (bv.w + cv.w);
    float mu = block_sum(x.x + x.y + x.z + x.w, red) * (1.0f / DMODEL);
    v4f d; d.x = x.x - mu; d.y = x.y - mu; d.z = x.z - mu; d.w = x.w - mu;
    float var = block_sum(d.x*d.x + d.y*d.y + d.z*d.z + d.w*d.w, red) * (1.0f / DMODEL);
    float rstd = rsqrtf(var + 1e-5f);
    v4f gg = *(const v4f*)(g + t * 4);
    v4f bb = *(const v4f*)(b + t * 4);
    v4f o;
    o.x = d.x * rstd * gg.x + bb.x;
    o.y = d.y * rstd * gg.y + bb.y;
    o.z = d.z * rstd * gg.z + bb.z;
    o.w = d.w * rstd * gg.w + bb.w;
    *(v4f*)(h + o4) = o;
}

// ---------------------------------------------------------------------------
// W2' prep: w2p[d, k*512+hh] = blk_W2[l,k,d,hh] * sigmoid(gate[l,k,d]+tor[d]) * excn[l,k]
// ---------------------------------------------------------------------------
__global__ __launch_bounds__(256)
void w2prep_kernel(const float* __restrict__ W2, const float* __restrict__ gateW,
                   const float* __restrict__ tor, const float* __restrict__ excn,
                   float* __restrict__ w2p, int l) {
    const size_t idx = (size_t)blockIdx.x * 256 + threadIdx.x; // 1024*2048 total
    const int kk = (int)(idx & 2047);
    const int d  = (int)(idx >> 11);
    const int k  = kk >> 9;
    const int hh = kk & 511;
    const float gt = sigmoidf_(gateW[(size_t)(l * NBLK + k) * DMODEL + d] + tor[d]);
    const float w  = W2[((size_t)(l * NBLK + k) * DMODEL + d) * 512 + hh];
    w2p[idx] = w * gt * excn[l * NBLK + k];
}

// ---------------------------------------------------------------------------
// WMMA GEMM: C[M,NN] = A[M,KK] (f16, row-major) * B[NN,KK]^T (f32 row-major,
// converted to f16 during LDS staging). Block tile 128x128, K-step 32,
// double-buffered LDS, 8 waves, each wave = 32x64 via 2x4
// v_wmma_f32_16x16x32_f16. NN/KK are compile-time so output stores use
// immediate offsets. EPI=0: f32 store. EPI=1: f16(gelu(x)) store.
// ---------------------------------------------------------------------------
template <int EPI, int NN, int KK>
__global__ __launch_bounds__(256)
void gemm_wmma(const _Float16* __restrict__ A, const float* __restrict__ Bw,
               float* __restrict__ C, _Float16* __restrict__ O16) {
    __shared__ _Float16 sA[2][128 * 40];   // stride 40 halves: 80B, 16B-aligned
    __shared__ _Float16 sB[2][128 * 40];

    const int tid  = threadIdx.x;
    const int bm   = blockIdx.y * 128;
    const int bn   = blockIdx.x * 128;
    const int wave = tid >> 5;
    const int lane = tid & 31;
    const int wm   = (wave >> 1) * 32;   // wave M offset: 0,32,64,96
    const int wn   = (wave & 1) * 64;    // wave N offset: 0,64
    const int l15  = lane & 15;
    const int lg   = lane >> 4;          // lane group 0/1

    // staging decomposition (256 threads cover 128x32 tiles)
    const int aRow0 = tid >> 2;          // 0..63  (chunk tid; +64 for chunk tid+256)
    const int aCol0 = (tid & 3) * 8;     // 0,8,16,24
    const int bRow0 = tid >> 3;          // 0..31  (+32*i for i=0..3)
    const int bCol0 = (tid & 7) * 4;     // 0..28

    const _Float16* aG = A  + (size_t)(bm + aRow0) * KK + aCol0;
    const float*    bG = Bw + (size_t)(bn + bRow0) * KK + bCol0;

    v8f acc[2][4];
    const v8f vz = {0.f, 0.f, 0.f, 0.f, 0.f, 0.f, 0.f, 0.f};
    #pragma unroll
    for (int i = 0; i < 2; ++i)
        #pragma unroll
        for (int j = 0; j < 4; ++j) acc[i][j] = vz;

    v4u aReg[2];
    v4f bReg[4];

    auto gload = [&](int kk) {
        aReg[0] = *(const v4u*)(aG + kk);
        aReg[1] = *(const v4u*)(aG + (size_t)64 * KK + kk);
        #pragma unroll
        for (int i = 0; i < 4; ++i)
            bReg[i] = *(const v4f*)(bG + (size_t)(i * 32) * KK + kk);
    };
    auto sstore = [&](int buf) {
        *(v4u*)(&sA[buf][aRow0 * 40 + aCol0])        = aReg[0];
        *(v4u*)(&sA[buf][(aRow0 + 64) * 40 + aCol0]) = aReg[1];
        #pragma unroll
        for (int i = 0; i < 4; ++i)
            *(v4h*)(&sB[buf][(bRow0 + i * 32) * 40 + bCol0]) =
                __builtin_convertvector(bReg[i], v4h);
    };

    // prologue: stage tile 0
    gload(0);
    sstore(0);
    __syncthreads();

    const int nsteps = KK / 32;
    for (int s = 0; s < nsteps; ++s) {
        const int cur = s & 1;
        if (s + 1 < nsteps) gload((s + 1) * 32);   // overlap with WMMAs below

        HFrag a[2], b[4];
        #pragma unroll
        for (int mt = 0; mt < 2; ++mt) {
            const _Float16* base = &sA[cur][(wm + mt * 16 + l15) * 40];
            a[mt].u[0] = *(const v4u*)(base + lg * 8);        // k = 8g..8g+7
            a[mt].u[1] = *(const v4u*)(base + 16 + lg * 8);   // k = 16+8g..23+8g
        }
        #pragma unroll
        for (int nt = 0; nt < 4; ++nt) {
            const _Float16* base = &sB[cur][(wn + nt * 16 + l15) * 40];
            b[nt].u[0] = *(const v4u*)(base + lg * 8);        // k = 8g..8g+7
            b[nt].u[1] = *(const v4u*)(base + 16 + lg * 8);   // k = 16+8g..23+8g
        }
        #pragma unroll
        for (int mt = 0; mt < 2; ++mt)
            #pragma unroll
            for (int nt = 0; nt < 4; ++nt)
                acc[mt][nt] = __builtin_amdgcn_wmma_f32_16x16x32_f16(
                    false, a[mt].h, false, b[nt].h, (short)0, acc[mt][nt], false, false);

        if (s + 1 < nsteps) sstore((s + 1) & 1);   // other buffer, no hazard
        __syncthreads();
    }

    // epilogue: C layout lane n = lane&15, row m = r + 8*lanegroup.
    // NN is compile-time -> stores use immediate row offsets.
    #pragma unroll
    for (int mt = 0; mt < 2; ++mt) {
        const int m0 = bm + wm + mt * 16 + lg * 8;
        #pragma unroll
        for (int nt = 0; nt < 4; ++nt) {
            const int n = bn + wn + nt * 16 + l15;
            if (EPI == 0) {
                float* p = C + (size_t)m0 * NN + n;
                #pragma unroll
                for (int r = 0; r < 8; ++r)
                    p[(size_t)r * NN] = acc[mt][nt][r];
            } else {
                _Float16* p = O16 + (size_t)m0 * NN + n;
                #pragma unroll
                for (int r = 0; r < 8; ++r)
                    p[(size_t)r * NN] = (_Float16)gelu_exact(acc[mt][nt][r]);
            }
        }
    }
}

// ---------------------------------------------------------------------------
// Host-side launch sequence
// ---------------------------------------------------------------------------
extern "C" void kernel_launch(void* const* d_in, const int* in_sizes, int n_in,
                              void* d_out, int out_size, void* d_ws, size_t ws_size,
                              hipStream_t stream) {
    (void)in_sizes; (void)n_in; (void)out_size; (void)ws_size;

    const int*   ids      = (const int*)  d_in[0];
    const float* emb      = (const float*)d_in[1];
    const float* pos      = (const float*)d_in[2];
    const float* torsion  = (const float*)d_in[3];
    const float* ln1_g    = (const float*)d_in[4];
    const float* ln1_b    = (const float*)d_in[5];
    const float* base_W   = (const float*)d_in[6];
    const float* blk_W1   = (const float*)d_in[7];
    const float* blk_W2   = (const float*)d_in[8];
    const float* blk_gate = (const float*)d_in[9];
    const float* prop_W   = (const float*)d_in[10];
    const float* prop_b   = (const float*)d_in[11];
    const float* ln2_g    = (const float*)d_in[12];
    const float* ln2_b    = (const float*)d_in[13];
    const float* oln_g    = (const float*)d_in[14];
    const float* oln_b    = (const float*)d_in[15];
    const float* lm_head  = (const float*)d_in[16];
    float* out = (float*)d_out;

    char* ws = (char*)d_ws;
    size_t off = 0;
    float*    excn = (float*)(ws + off);     off += 256;
    float*    h    = (float*)(ws + off);     off += (size_t)MROWS * DMODEL * 4;  // 8 MB
    float*    bbuf = (float*)(ws + off);     off += (size_t)MROWS * DMODEL * 4;  // 8 MB
    float*    cbuf = (float*)(ws + off);     off += (size_t)MROWS * DMODEL * 4;  // 8 MB
    float*    w2p  = (float*)(ws + off);     off += (size_t)DMODEL * 2048 * 4;   // 8 MB
    _Float16* hn16 = (_Float16*)(ws + off);  off += (size_t)MROWS * DMODEL * 2;  // 4 MB
    _Float16* t16  = (_Float16*)(ws + off);  off += (size_t)MROWS * 2048 * 2;    // 8 MB

    exc_kernel<<<1, 32, 0, stream>>>(prop_W, prop_b, excn);
    embed_kernel<<<MROWS, 256, 0, stream>>>(ids, emb, pos, h);

    for (int l = 0; l < NLAYER; ++l) {
        ln_to_f16_kernel<<<MROWS, 256, 0, stream>>>(h, ln1_g + l * DMODEL,
                                                    ln1_b + l * DMODEL, hn16);
        // base = hn @ base_W[l]^T   [2048 x 1024 x 1024]
        {
            dim3 grid(1024 / 128, MROWS / 128);
            gemm_wmma<0, 1024, 1024><<<grid, 256, 0, stream>>>(
                hn16, base_W + (size_t)l * DMODEL * DMODEL, bbuf, nullptr);
        }
        // t = gelu(hn @ W1stack^T)  [2048 x 2048 x 1024] -> f16
        {
            dim3 grid(2048 / 128, MROWS / 128);
            gemm_wmma<1, 2048, 1024><<<grid, 256, 0, stream>>>(
                hn16, blk_W1 + (size_t)l * 2048 * DMODEL, nullptr, t16);
        }
        // W2' = W2 * gate * excn (gate/excitement folded into weights)
        w2prep_kernel<<<(DMODEL * 2048) / 256, 256, 0, stream>>>(
            blk_W2, blk_gate, torsion, excn, w2p, l);
        // combined = t @ W2'^T      [2048 x 1024 x 2048]
        {
            dim3 grid(1024 / 128, MROWS / 128);
            gemm_wmma<0, 1024, 2048><<<grid, 256, 0, stream>>>(
                t16, w2p, cbuf, nullptr);
        }
        // h = LN2(h + 0.3*(base+combined))
        resid_ln_kernel<<<MROWS, 256, 0, stream>>>(h, bbuf, cbuf,
                                                   ln2_g + l * DMODEL,
                                                   ln2_b + l * DMODEL);
    }

    // final LN + lm head: logits = LN(h) @ lm_head^T  [2048 x 32000 x 1024]
    ln_to_f16_kernel<<<MROWS, 256, 0, stream>>>(h, oln_g, oln_b, hn16);
    {
        dim3 grid(32000 / 128, MROWS / 128);
        gemm_wmma<0, 32000, 1024><<<grid, 256, 0, stream>>>(hn16, lm_head, out, nullptr);
    }
}
